// FSUMGUCell_25469156065829
// MI455X (gfx1250) — compile-verified, matrix operand order checked by reading, not run on hardware
//
#include <hip/hip_runtime.h>
#include <cstdint>

// ---------------------------------------------------------------------------
// FSUMGUCell fused kernel for MI455X (gfx1250, wave32, WMMA).
//   fg = (W_f [hx;x] + b_f + 1)/2 ; ng = W_n [fg*hx;x] + b_n
//   hy = ng - fg*ng + fg*hx
// Per block of 64 batch rows: stage cat(hx,x) as bf16 in LDS, GEMM1 via
// v_wmma_f32_16x16x32_bf16 (f32 accum), overwrite hx region of the LDS
// A-buffer with bf16(fg*hx), GEMM2, fused epilogue. Weights pre-packed once
// into per-lane WMMA B-fragment order (bf16) in d_ws (L2-resident, 1.5 MB).
// K-sweep: 4 accumulator chains + explicit double-buffered A/B fragment
// prefetch so WMMAs overlap the next tile's loads (partial loadcnt waits).
// ---------------------------------------------------------------------------

typedef __attribute__((ext_vector_type(16))) __bf16 v16bf;
typedef __attribute__((ext_vector_type(8)))  float  v8f;

constexpr int kH   = 512;          // hidden
constexpr int kI   = 256;          // input
constexpr int kK   = 768;          // H + I (GEMM reduction dim)
constexpr int kKT  = kK / 32;      // 24 K-tiles of 32
constexpr int kNT  = kH / 16;      // 32 N-tiles of 16
constexpr int kM   = 64;           // batch rows per block
constexpr int kLDA = 776;          // LDS A row stride (bf16), padded for banks
constexpr int kLDF = 516;          // LDS fg row stride (f32), padded for banks
constexpr int kSMEM = kM * kLDA * 2 + kM * kLDF * 4;   // 231,424 B (<320 KB WGP)

static __device__ __forceinline__ unsigned short f2bf(float f) {
  unsigned int u = __float_as_uint(f);
  u += 0x7FFFu + ((u >> 16) & 1u);           // round-to-nearest-even
  return (unsigned short)(u >> 16);
}
static __device__ __forceinline__ unsigned int pk2(float lo, float hi) {
  return (unsigned int)f2bf(lo) | ((unsigned int)f2bf(hi) << 16);
}

union FragBF { v16bf v; uint4 q[2]; };

static __device__ __forceinline__ void loadA(FragBF& a, const unsigned short* ap) {
  a.q[0] = *(const uint4*)(ap);            // K chunk {0..7}+delta
  a.q[1] = *(const uint4*)(ap + 16);       // K chunk {16..23}+delta
}
static __device__ __forceinline__ void loadB(FragBF& b, const unsigned short* p) {
  b.q[0] = *(const uint4*)(p);
  b.q[1] = *(const uint4*)(p + 8);
}

// --- pack f32 weight [512 x 768] row-major into bf16 WMMA B-fragments ------
// B = W^T (32x16 per tile, 16-bit operands, wave32):
//   lane l (<16): column n = l,     K offsets {0..7} U {16..23}
//   lane l (>=16): column n = l-16, K offsets {8..15} U {24..31}
// Per (n_tile, k_tile) block: 32 lanes x 16 bf16, lane-contiguous (32 B/lane).
__global__ __launch_bounds__(256) void pack_w(const float* __restrict__ w,
                                              unsigned short* __restrict__ wpk) {
  int idx = blockIdx.x * 256 + threadIdx.x;
  if (idx >= kNT * kKT * 32) return;
  int lane = idx & 31;
  int t    = idx >> 5;                 // tile id = n_tile*kKT + k_tile
  int kt   = t % kKT;
  int nt   = t / kKT;
  int n    = nt * 16 + (lane & 15);
  int kb   = kt * 32 + ((lane >> 4) << 3);
  const float* src = w + (size_t)n * kK + kb;      // B[k,n] = w[n,k]: row of w
  unsigned int* dst = (unsigned int*)(wpk + (size_t)idx * 16);
#pragma unroll
  for (int j = 0; j < 4; ++j) dst[j]     = pk2(src[2 * j],      src[2 * j + 1]);
#pragma unroll
  for (int j = 0; j < 4; ++j) dst[4 + j] = pk2(src[16 + 2 * j], src[16 + 2 * j + 1]);
}

// K-sweep for four adjacent N-tiles sharing one A fragment, with explicit
// double-buffered prefetch: issue tile kt+1's loads before tile kt's WMMAs.
static __device__ __forceinline__ void gemm4(const unsigned short* sArow,
                                             const unsigned short* wp,
                                             v8f acc[4]) {
  FragBF a[2];
  FragBF b[2][4];
  loadA(a[0], sArow);
#pragma unroll
  for (int t = 0; t < 4; ++t) loadB(b[0][t], wp + t * (kKT * 512));

#pragma unroll 2
  for (int kt = 0; kt < kKT; ++kt) {
    const int cur = kt & 1;
    const int nxt = cur ^ 1;
    const int ktp = (kt + 1 < kKT) ? (kt + 1) : kt;   // clamped prefetch idx
    // prefetch next tile's fragments into the other buffer
    loadA(a[nxt], sArow + ktp * 32);
#pragma unroll
    for (int t = 0; t < 4; ++t)
      loadB(b[nxt][t], wp + t * (kKT * 512) + ktp * 512);
    // consume current tile (waits only on loads issued last iteration)
#pragma unroll
    for (int t = 0; t < 4; ++t)
      acc[t] = __builtin_amdgcn_wmma_f32_16x16x32_bf16(
          false, a[cur].v, false, b[cur][t].v, (short)0, acc[t], false, false);
  }
}

__global__ __launch_bounds__(256) void fsumgu_main(
    const float* __restrict__ x, const float* __restrict__ hx,
    const unsigned short* __restrict__ wpkF, const float* __restrict__ bF,
    const unsigned short* __restrict__ wpkN, const float* __restrict__ bN,
    float* __restrict__ out) {
  extern __shared__ char smem[];
  unsigned short* sA  = (unsigned short*)smem;            // [64][776] bf16
  float*          sFG = (float*)(smem + kM * kLDA * 2);   // [64][516] f32

  const int tid  = threadIdx.x;
  const int lane = tid & 31;
  const int wave = tid >> 5;
  const int l15  = lane & 15;
  const int hi16 = lane >> 4;
  const int m0   = (wave & 3) * 16;   // 4 row-groups of 16
  const int nh   = wave >> 2;         // 2 N-halves of 256
  const size_t rowBase = (size_t)blockIdx.x * kM;

  // ---- stage cat(hx, x) rows -> bf16 LDS A-buffer (coalesced float4) ----
  for (int i = tid; i < kM * (kK / 4); i += 256) {
    int r  = i / (kK / 4);
    int c4 = (i % (kK / 4)) * 4;
    float4 v;
    if (c4 < kH) v = *(const float4*)&hx[(rowBase + r) * kH + c4];
    else         v = *(const float4*)&x[(rowBase + r) * kI + (c4 - kH)];
    unsigned int* d = (unsigned int*)&sA[(size_t)r * kLDA + c4];
    d[0] = pk2(v.x, v.y);
    d[1] = pk2(v.z, v.w);
  }
  __syncthreads();

  // A fragment base for this lane: row m0+l15, K-delta 8 for upper lanes.
  const unsigned short* sArow = sA + (size_t)(m0 + l15) * kLDA + (hi16 << 3);
  const int mTop = m0 + (hi16 << 3);   // C/D layout: vgpr r -> M = mTop + r

  // ---- stage 1: fg = (W_f [hx;x] + b_f + 1)/2, kept f32 in LDS ----
  for (int g = 0; g < 4; ++g) {
    int nt0 = nh * 16 + g * 4;
    v8f acc[4] = {{}, {}, {}, {}};
    gemm4(sArow, wpkF + ((size_t)nt0 * kKT * 32 + lane) * 16, acc);
#pragma unroll
    for (int t = 0; t < 4; ++t) {
      int n = (nt0 + t) * 16 + l15;
      float bias = bF[n];
#pragma unroll
      for (int r = 0; r < 8; ++r) {
        float fg = (acc[t][r] + bias + 1.0f) * 0.5f;
        sFG[(size_t)(mTop + r) * kLDF + n] = fg;
      }
    }
  }
  __syncthreads();   // all waves done reading hx region of sA

  // ---- overwrite A[:, 0:512] with bf16(fg * hx)  (coalesced) ----
  for (int i = tid; i < kM * (kH / 4); i += 256) {
    int r  = i / (kH / 4);
    int c4 = (i % (kH / 4)) * 4;
    float4 h = *(const float4*)&hx[(rowBase + r) * kH + c4];
    const float* f = &sFG[(size_t)r * kLDF + c4];
    unsigned int* d = (unsigned int*)&sA[(size_t)r * kLDA + c4];
    d[0] = pk2(f[0] * h.x, f[1] * h.y);
    d[1] = pk2(f[2] * h.z, f[3] * h.w);
  }
  __syncthreads();

  // ---- stage 2: ng = W_n [fg*hx;x] + b_n ; hy = ng*(1-fg) + fg*hx ----
  for (int g = 0; g < 4; ++g) {
    int nt0 = nh * 16 + g * 4;
    v8f acc[4] = {{}, {}, {}, {}};
    gemm4(sArow, wpkN + ((size_t)nt0 * kKT * 32 + lane) * 16, acc);
#pragma unroll
    for (int t = 0; t < 4; ++t) {
      int n = (nt0 + t) * 16 + l15;
      float bias = bN[n];
#pragma unroll
      for (int r = 0; r < 8; ++r) {
        int m = mTop + r;
        size_t gr = rowBase + m;
        float ng = acc[t][r] + bias;
        float fg = sFG[(size_t)m * kLDF + n];
        float hv = hx[gr * kH + n];           // L2-hot reload, 64B segments
        out[gr * kH + n] = ng - fg * ng + fg * hv;
      }
    }
  }
}

extern "C" void kernel_launch(void* const* d_in, const int* in_sizes, int n_in,
                              void* d_out, int out_size, void* d_ws, size_t ws_size,
                              hipStream_t stream) {
  const float* x   = (const float*)d_in[0];
  const float* hxp = (const float*)d_in[1];
  const float* wf  = (const float*)d_in[2];
  const float* bf  = (const float*)d_in[3];
  const float* wn  = (const float*)d_in[4];
  const float* bn  = (const float*)d_in[5];
  float* out = (float*)d_out;

  // d_ws: two bf16-packed weight fragment arrays (768 KB each).
  unsigned short* wpkF = (unsigned short*)d_ws;
  unsigned short* wpkN = wpkF + (size_t)kH * kK;

  const int packThreads = kNT * kKT * 32;   // 24576
  pack_w<<<(packThreads + 255) / 256, 256, 0, stream>>>(wf, wpkF);
  pack_w<<<(packThreads + 255) / 256, 256, 0, stream>>>(wn, wpkN);

  const int B = in_sizes[1] / kH;           // 65536
  fsumgu_main<<<B / kM, 256, kSMEM, stream>>>(x, hxp, wpkF, bf, wpkN, bn, out);
}
